// fancy_mlp_12395275616798
// MI455X (gfx1250) — matrix operand order repaired and checked by reading, not compiled, not run on hardware
//
#include <hip/hip_runtime.h>
#include <hip/hip_bf16.h>
#include <math.h>

typedef __attribute__((ext_vector_type(16))) _Float16 v16h;
typedef __attribute__((ext_vector_type(8)))  _Float16 v8h;
typedef __attribute__((ext_vector_type(4)))  _Float16 v4h;
typedef __attribute__((ext_vector_type(8)))  float    v8f;

#define DD 20
#define NBLOCKS 1024

__global__ void init_ws_kernel(float* ws) {
    ws[0] = 0.0f;
    ws[1] = 0.0f;
}

// Layouts (wave32):
//  A (16x32 f16): lane half hi=0 -> K blocks {0..7, 16..23}, hi=1 -> {8..15, 24..31};
//                 a[2v],a[2v+1] = K = (v>=4)*16 + hi*8 + (v&3)*2 (+1)
//  B (32x16 f16): same K order as A; b[2v],b[2v+1] = K = (v>=4)*16 + hi*8 + (v&3)*2 (+1)
//  C/D (16x16 f32): vgpr v -> row v + hi*8, col = lane&15 (+16 for N-tile 1)
// Bias trick: virtual row K=20 (A[m][20]=1 in a[12] of hi=0 lanes; B row 20 in b[12] of
// hi=0 lanes) folds "+ b" / "+ 1.0" into the WMMA itself.

__global__ void __launch_bounds__(256) fancy_mlp_main(
    const float* __restrict__ x, const float* __restrict__ W,
    const float* __restrict__ b, const float* __restrict__ R,
    float* __restrict__ ws, int nrows, int mode)
{
    __shared__ _Float16 stage[8][16][40];   // per-wave 16x32 tile, stride 40 (cols 32..39 pad)
    __shared__ float rs[256];
    __shared__ float rq[256];

    const int tid  = threadIdx.x;
    const int lane = tid & 31;
    const int wave = tid >> 5;
    const int hi   = lane >> 4;
    const int nl   = lane & 15;

    // zero the staging pad cols [32..35] once (read as "zeros" by hi=1 A-rebuild)
    if (lane < 16) {
        v4h z = {};
        *(v4h*)&stage[wave][nl][32] = z;
    }

    // ---- loop-invariant B operands (Wt[k][n] = W[n][k], R[k][n]), bias row at K=20 ----
    v16h bW0, bW1, bR0, bR1;
#pragma unroll
    for (int v = 0; v < 8; ++v) {
        const int k0 = (v >= 4) * 16 + hi * 8 + (v & 3) * 2;
        const int k1 = k0 + 1;
        const int n0 = nl;
        const int n1 = nl + 16;
        bW0[2*v]   = (k0 < DD) ? (_Float16)W[n0 * DD + k0] : (_Float16)0.0f;
        bW0[2*v+1] = (k1 < DD) ? (_Float16)W[n0 * DD + k1] : (_Float16)0.0f;
        bW1[2*v]   = (k0 < DD && n1 < DD) ? (_Float16)W[n1 * DD + k0] : (_Float16)0.0f;
        bW1[2*v+1] = (k1 < DD && n1 < DD) ? (_Float16)W[n1 * DD + k1] : (_Float16)0.0f;
        bR0[2*v]   = (k0 < DD) ? (_Float16)R[k0 * DD + n0] : (_Float16)0.0f;
        bR0[2*v+1] = (k1 < DD) ? (_Float16)R[k1 * DD + n0] : (_Float16)0.0f;
        bR1[2*v]   = (k0 < DD && n1 < DD) ? (_Float16)R[k0 * DD + n1] : (_Float16)0.0f;
        bR1[2*v+1] = (k1 < DD && n1 < DD) ? (_Float16)R[k1 * DD + n1] : (_Float16)0.0f;
    }
    // virtual bias row K=20 lives in b[12] (hi=0, v=6, k0=20)
    if (hi == 0) {
        bW0[12] = (_Float16)b[nl];
        bW1[12] = (nl + 16 < DD) ? (_Float16)b[nl + 16] : (_Float16)0.0f;
        bR0[12] = (_Float16)1.0f;                                  // the "+ 1.0" of layer 2
        bR1[12] = (nl + 16 < DD) ? (_Float16)1.0f : (_Float16)0.0f;
    }
    const _Float16 oneA = (hi == 0) ? (_Float16)1.0f : (_Float16)0.0f;  // A[m][20] = 1

    v8f accS8 = {}, accQ8 = {};

    const int ntiles = nrows >> 4;            // 2M % 16 == 0
    const int gwave  = blockIdx.x * 8 + wave;
    const int nwaves = gridDim.x * 8;

    for (int t = gwave; t < ntiles; t += nwaves) {
        const float* xr = x + (size_t)(t * 16 + nl) * DD;   // lane's A-row (row = t*16 + nl)

        // uniform-clamped prefetch of next tile
        {
            int tn = t + nwaves;
            if (tn >= ntiles) tn = 0;
            __builtin_prefetch((const void*)(x + (size_t)(tn * 16 + nl) * DD), 0, 0);
        }

        // ---- A1 from x: rows are 16B-aligned (stride 80B) -> 3 x float4 loads ----
        const float4 f0 = *(const float4*)(xr + hi * 8);        // K = hi*8 + 0..3
        const float4 f1 = *(const float4*)(xr + hi * 8 + 4);    // K = hi*8 + 4..7
        const float4 f2 = *(const float4*)(xr + 16);            // K = 16..19 (hi=0 only; coalesces)
        v16h a;
        a[0] = (_Float16)f0.x; a[1] = (_Float16)f0.y;
        a[2] = (_Float16)f0.z; a[3] = (_Float16)f0.w;
        a[4] = (_Float16)f1.x; a[5] = (_Float16)f1.y;
        a[6] = (_Float16)f1.z; a[7] = (_Float16)f1.w;
        a[8]  = hi ? (_Float16)0.0f : (_Float16)f2.x;
        a[9]  = hi ? (_Float16)0.0f : (_Float16)f2.y;
        a[10] = hi ? (_Float16)0.0f : (_Float16)f2.z;
        a[11] = hi ? (_Float16)0.0f : (_Float16)f2.w;
        a[12] = oneA;                      // K=20 bias slot
        a[13] = (_Float16)0.0f; a[14] = (_Float16)0.0f; a[15] = (_Float16)0.0f;

        // ---- layer 1: h1 = x @ W^T + b (bias folded) ----
        v8f c0 = {}, c1 = {};
        c0 = __builtin_amdgcn_wmma_f32_16x16x32_f16(false, a, false, bW0, (short)0, c0, false, false);
        c1 = __builtin_amdgcn_wmma_f32_16x16x32_f16(false, a, false, bW1, (short)0, c1, false, false);

        // ---- stage (D-layout) -> rebuild A2 (wave-private LDS, in-order per wave) ----
#pragma unroll
        for (int v = 0; v < 8; ++v) {
            const int m = v + hi * 8;
            stage[wave][m][nl]      = (_Float16)c0[v];
            stage[wave][m][nl + 16] = (_Float16)c1[v];
        }
        asm volatile("" ::: "memory");
        {
            v8h p = *(const v8h*)&stage[wave][nl][hi * 8];        // K 0..7 / 8..15
            v4h q = *(const v4h*)&stage[wave][nl][16 + hi * 16];  // K 16..19 / zeros
#pragma unroll
            for (int i = 0; i < 8; ++i) a[i] = p[i];
#pragma unroll
            for (int i = 0; i < 4; ++i) a[8 + i] = q[i];
            a[12] = oneA;
            a[13] = (_Float16)0.0f; a[14] = (_Float16)0.0f; a[15] = (_Float16)0.0f;
        }

        // ---- layer 2: h2 = relu(h1 @ R + 1) ("+1" folded via bias row) ----
        v8f d0 = {}, d1 = {};
        d0 = __builtin_amdgcn_wmma_f32_16x16x32_f16(false, a, false, bR0, (short)0, d0, false, false);
        d1 = __builtin_amdgcn_wmma_f32_16x16x32_f16(false, a, false, bR1, (short)0, d1, false, false);

        // ---- stage relu(h2) -> rebuild A3 ----
#pragma unroll
        for (int v = 0; v < 8; ++v) {
            const int m = v + hi * 8;
            stage[wave][m][nl]      = (_Float16)fmaxf(d0[v], 0.0f);
            stage[wave][m][nl + 16] = (_Float16)fmaxf(d1[v], 0.0f);
        }
        asm volatile("" ::: "memory");
        {
            v8h p = *(const v8h*)&stage[wave][nl][hi * 8];
            v4h q = *(const v4h*)&stage[wave][nl][16 + hi * 16];
#pragma unroll
            for (int i = 0; i < 8; ++i) a[i] = p[i];
#pragma unroll
            for (int i = 0; i < 4; ++i) a[8 + i] = q[i];
            a[12] = oneA;
            a[13] = (_Float16)0.0f; a[14] = (_Float16)0.0f; a[15] = (_Float16)0.0f;
        }

        // ---- layer 3: h3 = h2 @ W^T + b (bias folded) ----
        v8f e0 = {}, e1 = {};
        e0 = __builtin_amdgcn_wmma_f32_16x16x32_f16(false, a, false, bW0, (short)0, e0, false, false);
        e1 = __builtin_amdgcn_wmma_f32_16x16x32_f16(false, a, false, bW1, (short)0, e1, false, false);

        // ---- vector accumulate (padded cols are exactly 0) ----
#pragma unroll
        for (int v = 0; v < 8; ++v) {
            accS8[v] += e0[v] + e1[v];
            accQ8[v] += e0[v] * e0[v] + e1[v] * e1[v];
        }
    }

    // horizontal reduce of the v8f accumulators
    float accS = 0.0f, accQ = 0.0f;
#pragma unroll
    for (int v = 0; v < 8; ++v) { accS += accS8[v]; accQ += accQ8[v]; }

    // ---- block reduction ----
    rs[tid] = accS;
    rq[tid] = accQ;
    __syncthreads();
#pragma unroll
    for (int s = 128; s > 0; s >>= 1) {
        if (tid < s) { rs[tid] += rs[tid + s]; rq[tid] += rq[tid + s]; }
        __syncthreads();
    }
    if (tid == 0) {
        if (mode == 1) {                      // deterministic per-block partials
            ws[2 + 2 * blockIdx.x]     = rs[0];
            ws[2 + 2 * blockIdx.x + 1] = rq[0];
        } else {                              // fallback: atomics
            atomicAdd(&ws[0], rs[0]);
            atomicAdd(&ws[1], rq[0]);
        }
    }
}

__global__ void finalize_kernel(const float* __restrict__ ws, float* __restrict__ out,
                                int mode, int nblocks) {
    float s, q;
    if (mode == 1) {
        s = 0.0f; q = 0.0f;
        for (int i = 0; i < nblocks; ++i) {   // serial => bitwise deterministic
            s += ws[2 + 2 * i];
            q += ws[2 + 2 * i + 1];
        }
    } else {
        s = ws[0]; q = ws[1];
    }
    float n = sqrtf(q);                        // ||h3||
    int k = 0;
    while (n > 1.0f && k < 256) { n *= 0.5f; ++k; }   // exact halving loop
    const float scale = exp2f((float)(-k));
    const float mult  = (n < 0.8f) ? 10.0f : 1.0f;    // n == ||h3 * 2^-k||
    out[0] = s * scale * mult;
}

extern "C" void kernel_launch(void* const* d_in, const int* in_sizes, int n_in,
                              void* d_out, int out_size, void* d_ws, size_t ws_size,
                              hipStream_t stream) {
    const float* x = (const float*)d_in[0];
    const float* W = (const float*)d_in[1];
    const float* b = (const float*)d_in[2];
    const float* R = (const float*)d_in[3];
    float* ws  = (float*)d_ws;
    float* out = (float*)d_out;
    const int nrows = in_sizes[0] / DD;

    const int mode = (ws_size >= (size_t)(2 + 2 * NBLOCKS) * sizeof(float)) ? 1 : 0;

    init_ws_kernel<<<1, 1, 0, stream>>>(ws);
    fancy_mlp_main<<<NBLOCKS, 256, 0, stream>>>(x, W, b, R, ws, nrows, mode);
    finalize_kernel<<<1, 1, 0, stream>>>(ws, out, mode, NBLOCKS);
}
